// DLRM_5403068858958
// MI455X (gfx1250) — matrix operand order repaired and checked by reading, not compiled, not run on hardware
//
#include <hip/hip_runtime.h>
#include <hip/hip_bf16.h>

typedef _Float16 v16h __attribute__((ext_vector_type(16)));
typedef _Float16 h8   __attribute__((ext_vector_type(8)));
typedef float    v8f  __attribute__((ext_vector_type(8)));

#define BN_INV 0.9999950000374997f

// ---------------------------------------------------------------------------
// Prep kernels
// ---------------------------------------------------------------------------

// continuous [B][13] f32 -> [B][32] f16 (zero padded K)
__global__ void prep_x(const float* __restrict__ x, _Float16* __restrict__ xf,
                       int B) {
  int i = blockIdx.x * blockDim.x + threadIdx.x;
  if (i >= B * 32) return;
  int b = i >> 5, c = i & 31;
  xf[i] = (c < 13) ? (_Float16)x[b * 13 + c] : (_Float16)0.f;
}

// W [K][N] f32 -> Wt [N][Kp] f16 (zero padded K..Kp)
__global__ void prep_wt(const float* __restrict__ W, _Float16* __restrict__ Wt,
                        int K, int Kp, int N) {
  int i = blockIdx.x * blockDim.x + threadIdx.x;
  if (i >= N * Kp) return;
  int n = i / Kp, k = i - n * Kp;
  Wt[(size_t)n * Kp + k] = (k < K) ? (_Float16)W[(size_t)k * N + n]
                                   : (_Float16)0.f;
}

// fold bias + batchnorm affine into per-channel scale/shift
__global__ void prep_st(const float* __restrict__ g, const float* __restrict__ be,
                        const float* __restrict__ bias,
                        float* __restrict__ s, float* __restrict__ t,
                        int N, int has_bn) {
  int i = blockIdx.x * blockDim.x + threadIdx.x;
  if (i >= N) return;
  if (has_bn) {
    float sc = g[i] * BN_INV;
    s[i] = sc;
    t[i] = bias[i] * sc + be[i];
  } else {
    s[i] = 1.f;
    t[i] = bias[i];
  }
}

// ---------------------------------------------------------------------------
// Generic fused GEMM + scale/shift (+ReLU) with WMMA f16->f32.
//   A  : [M][lda] f16 row-major activations
//   Wt : [N][ldw] f16 (transposed weights, K contiguous)
//   Out: [M][ldo] f16 = act( (A@W) * s[n] + t[n] )
// BM=128, BK=32, 8 waves; wave w computes row-tile w across BN_TILES n-tiles.
// Register-staged double-buffered pipeline: global loads for tile k+1 are in
// flight across the barrier and the BN_TILES WMMAs of tile k.
// ---------------------------------------------------------------------------
template <int BN_TILES>
__global__ __launch_bounds__(256) void gemm_wmma(
    const _Float16* __restrict__ A, int lda,
    const _Float16* __restrict__ Wt, int ldw,
    const float* __restrict__ s, const float* __restrict__ t,
    _Float16* __restrict__ Out, int ldo, int K, int relu) {
  constexpr int BM = 128;
  constexpr int BN = BN_TILES * 16;
  constexpr int PAD = 40;              // halves per LDS row (bank padding)
  constexpr int ACH = (BM * 4) / 256;  // 16B A-chunks per thread (=2)
  constexpr int BCH = (BN * 4) / 256;  // 16B B-chunks per thread (2 or 1)
  __shared__ __attribute__((aligned(16))) _Float16 sA[2][BM * PAD];
  __shared__ __attribute__((aligned(16))) _Float16 sB[2][BN * PAD];

  const int tid = threadIdx.x;
  const int wave = tid >> 5;
  const int lane = tid & 31;
  const int row16 = lane & 15;
  const int hi = lane >> 4;
  const size_t m0 = (size_t)blockIdx.y * BM;
  const int n0 = blockIdx.x * BN;

  // static chunk coordinates (no exec-masked copy loops)
  int arow[ACH], acol[ACH], brow[BCH], bcol[BCH];
#pragma unroll
  for (int i = 0; i < ACH; ++i) {
    int c = tid + i * 256;
    arow[i] = c >> 2;
    acol[i] = (c & 3) * 8;
  }
#pragma unroll
  for (int i = 0; i < BCH; ++i) {
    int c = tid + i * 256;
    brow[i] = c >> 2;
    bcol[i] = (c & 3) * 8;
  }

  v8f acc[BN_TILES];
#pragma unroll
  for (int i = 0; i < BN_TILES; ++i)
#pragma unroll
    for (int j = 0; j < 8; ++j) acc[i][j] = 0.f;

  // prologue: fetch tile 0 into registers
  h8 ra[ACH], rb[BCH];
#pragma unroll
  for (int i = 0; i < ACH; ++i)
    ra[i] = *(const h8*)(A + (m0 + arow[i]) * (size_t)lda + acol[i]);
#pragma unroll
  for (int i = 0; i < BCH; ++i)
    rb[i] = *(const h8*)(Wt + (size_t)(n0 + brow[i]) * ldw + bcol[i]);

  int buf = 0;
  for (int k0 = 0; k0 < K; k0 += 32) {
    // commit current registers to LDS buffer `buf`
#pragma unroll
    for (int i = 0; i < ACH; ++i)
      *(h8*)&sA[buf][arow[i] * PAD + acol[i]] = ra[i];
#pragma unroll
    for (int i = 0; i < BCH; ++i)
      *(h8*)&sB[buf][brow[i] * PAD + bcol[i]] = rb[i];

    // issue global loads for the next tile (overlap with barrier + WMMAs)
    const bool more = (k0 + 32) < K;
    if (more) {
      const int kn = k0 + 32;
#pragma unroll
      for (int i = 0; i < ACH; ++i)
        ra[i] = *(const h8*)(A + (m0 + arow[i]) * (size_t)lda + kn + acol[i]);
#pragma unroll
      for (int i = 0; i < BCH; ++i)
        rb[i] = *(const h8*)(Wt + (size_t)(n0 + brow[i]) * ldw + kn + bcol[i]);
    }
    __syncthreads();

    // A fragment: lane-lo holds K {0..7, 16..23}, lane-hi K {8..15, 24..31}
    const _Float16* ap = &sA[buf][(wave * 16 + row16) * PAD + hi * 8];
    h8 alo = *(const h8*)ap;
    h8 ahi = *(const h8*)(ap + 16);
    v16h af;
#pragma unroll
    for (int i = 0; i < 8; ++i) { af[i] = alo[i]; af[8 + i] = ahi[i]; }

#pragma unroll
    for (int nt = 0; nt < BN_TILES; ++nt) {
      // B fragment: lane-lo holds K {0..7, 16..23}, lane-hi K {8..15, 24..31}
      const _Float16* bp = &sB[buf][(nt * 16 + row16) * PAD + hi * 8];
      h8 b0 = *(const h8*)bp;
      h8 b1 = *(const h8*)(bp + 16);
      v16h bf;
#pragma unroll
      for (int i = 0; i < 8; ++i) { bf[i] = b0[i]; bf[8 + i] = b1[i]; }
      acc[nt] = __builtin_amdgcn_wmma_f32_16x16x32_f16(
          false, af, false, bf, (short)0, acc[nt], false, false);
    }
    buf ^= 1;
  }

  // epilogue: scale/shift (+relu), f16 store.  C layout: VGPR r -> M=r+8*hi.
#pragma unroll
  for (int nt = 0; nt < BN_TILES; ++nt) {
    int n = n0 + nt * 16 + row16;
    float sc = s[n], sh = t[n];
#pragma unroll
    for (int r = 0; r < 8; ++r) {
      float v = acc[nt][r] * sc + sh;
      if (relu && v < 0.f) v = 0.f;
      Out[(m0 + wave * 16 + hi * 8 + r) * (size_t)ldo + n] = (_Float16)v;
    }
  }
}

// ---------------------------------------------------------------------------
// Embedding gather: write f16 rows straight into top_in concat region
// top_in layout per row (2080 f16): [0..350]=inter, [351..2078]=all_emb, [2079]=0
// ---------------------------------------------------------------------------
__global__ void gather_emb(const float* __restrict__ emb,
                           const int* __restrict__ idx,
                           _Float16* __restrict__ top_in) {
  int g = blockIdx.x * 256 + threadIdx.x;  // B*26*64 threads exactly
  int d = g & 63;
  int f = (g >> 6) % 26;
  int b = g / (26 * 64);
  int id = idx[b * 26 + f];
  float v = emb[((size_t)f * 100000 + id) * 64 + d];
  top_in[(size_t)b * 2080 + 351 + (f + 1) * 64 + d] = (_Float16)v;
}

// ---------------------------------------------------------------------------
// Pairwise interactions: per sample 27x64 gram upper triangle (351 dots)
// 8 samples per block (one per wave), tile staged in LDS.
// ---------------------------------------------------------------------------
__global__ __launch_bounds__(256) void interact(_Float16* __restrict__ top_in) {
  __shared__ _Float16 e[8][27 * 64];
  int wave = threadIdx.x >> 5, lane = threadIdx.x & 31;
  size_t base = ((size_t)blockIdx.x * 8 + wave) * 2080;
  for (int i = lane; i < 27 * 64; i += 32) e[wave][i] = top_in[base + 351 + i];
  __syncthreads();
  for (int p = lane; p < 352; p += 32) {
    if (p == 351) {  // zero the K-pad column
      top_in[base + 2079] = (_Float16)0.f;
      continue;
    }
    int i = 0, rem = p;
    while (rem >= 26 - i) { rem -= 26 - i; ++i; }
    int j = i + 1 + rem;
    float acc = 0.f;
#pragma unroll 8
    for (int d = 0; d < 64; ++d)
      acc += (float)e[wave][i * 64 + d] * (float)e[wave][j * 64 + d];
    top_in[base + p] = (_Float16)acc;
  }
}

// ---------------------------------------------------------------------------
// Final 128 -> 1 layer (tiny, f32)
// ---------------------------------------------------------------------------
__global__ void final_layer(const _Float16* __restrict__ h6,
                            const float* __restrict__ W7,
                            const float* __restrict__ b7,
                            float* __restrict__ out) {
  int b = blockIdx.x * blockDim.x + threadIdx.x;
  float acc = b7[0];
#pragma unroll 8
  for (int d = 0; d < 128; ++d) acc += (float)h6[(size_t)b * 128 + d] * W7[d];
  out[b] = acc;
}

// ---------------------------------------------------------------------------
// Host launcher
// ---------------------------------------------------------------------------
extern "C" void kernel_launch(void* const* d_in, const int* in_sizes, int n_in,
                              void* d_out, int out_size, void* d_ws,
                              size_t ws_size, hipStream_t stream) {
  (void)in_sizes; (void)n_in; (void)out_size; (void)ws_size;
  const int B = 16384;

  const float* continuous = (const float*)d_in[0];
  const int*   cat_idx    = (const int*)d_in[1];
  const float* emb_tables = (const float*)d_in[2];
  const float* W1 = (const float*)d_in[3],  *b1 = (const float*)d_in[4];
  const float* g1 = (const float*)d_in[5],  *be1 = (const float*)d_in[6];
  const float* W2 = (const float*)d_in[7],  *b2 = (const float*)d_in[8];
  const float* g2 = (const float*)d_in[9],  *be2 = (const float*)d_in[10];
  const float* W3 = (const float*)d_in[11], *b3 = (const float*)d_in[12];
  const float* W4 = (const float*)d_in[13], *b4 = (const float*)d_in[14];
  const float* g4 = (const float*)d_in[15], *be4 = (const float*)d_in[16];
  const float* W5 = (const float*)d_in[17], *b5 = (const float*)d_in[18];
  const float* g5 = (const float*)d_in[19], *be5 = (const float*)d_in[20];
  const float* W6 = (const float*)d_in[21], *b6 = (const float*)d_in[22];
  const float* g6 = (const float*)d_in[23], *be6 = (const float*)d_in[24];
  const float* W7 = (const float*)d_in[25], *b7 = (const float*)d_in[26];

  // workspace carve (256B aligned slices)
  unsigned char* w = (unsigned char*)d_ws;
  auto carve = [&](size_t bytes) -> void* {
    void* p = (void*)w;
    w += (bytes + 255) & ~(size_t)255;
    return p;
  };
  _Float16* xf16  = (_Float16*)carve((size_t)B * 32 * 2);
  _Float16* W1t   = (_Float16*)carve((size_t)256 * 32 * 2);
  _Float16* W2t   = (_Float16*)carve((size_t)128 * 256 * 2);
  _Float16* W3t   = (_Float16*)carve((size_t)64 * 128 * 2);
  _Float16* W4t   = (_Float16*)carve((size_t)512 * 2080 * 2);
  _Float16* W5t   = (_Float16*)carve((size_t)256 * 512 * 2);
  _Float16* W6t   = (_Float16*)carve((size_t)128 * 256 * 2);
  float*    stbuf = (float*)carve(2688 * 4);
  _Float16* top_in = (_Float16*)carve((size_t)B * 2080 * 2);
  _Float16* h1    = (_Float16*)carve((size_t)B * 256 * 2);  // reused as h5
  _Float16* h2    = (_Float16*)carve((size_t)B * 128 * 2);  // reused as h6
  _Float16* h4    = (_Float16*)carve((size_t)B * 512 * 2);
  _Float16* h5 = h1;
  _Float16* h6 = h2;

  float *s1 = stbuf + 0,    *t1 = stbuf + 256;
  float *s2 = stbuf + 512,  *t2 = stbuf + 640;
  float *s3 = stbuf + 768,  *t3 = stbuf + 832;
  float *s4 = stbuf + 896,  *t4 = stbuf + 1408;
  float *s5 = stbuf + 1920, *t5 = stbuf + 2176;
  float *s6 = stbuf + 2432, *t6 = stbuf + 2560;

  // ---- prep ----
  prep_x<<<(B * 32) / 256, 256, 0, stream>>>(continuous, xf16, B);
  prep_wt<<<(256 * 32) / 256, 256, 0, stream>>>(W1, W1t, 13, 32, 256);
  prep_wt<<<(128 * 256) / 256, 256, 0, stream>>>(W2, W2t, 256, 256, 128);
  prep_wt<<<(64 * 128) / 256, 256, 0, stream>>>(W3, W3t, 128, 128, 64);
  prep_wt<<<(512 * 2080 + 255) / 256, 256, 0, stream>>>(W4, W4t, 2079, 2080, 512);
  prep_wt<<<(256 * 512) / 256, 256, 0, stream>>>(W5, W5t, 512, 512, 256);
  prep_wt<<<(128 * 256) / 256, 256, 0, stream>>>(W6, W6t, 256, 256, 128);
  prep_st<<<1, 256, 0, stream>>>(g1, be1, b1, s1, t1, 256, 1);
  prep_st<<<1, 256, 0, stream>>>(g2, be2, b2, s2, t2, 128, 1);
  prep_st<<<1, 256, 0, stream>>>(b3, b3, b3, s3, t3, 64, 0);
  prep_st<<<2, 256, 0, stream>>>(g4, be4, b4, s4, t4, 512, 1);
  prep_st<<<1, 256, 0, stream>>>(g5, be5, b5, s5, t5, 256, 1);
  prep_st<<<1, 256, 0, stream>>>(g6, be6, b6, s6, t6, 128, 1);

  // ---- embedding gather (independent of bottom MLP) ----
  gather_emb<<<(B * 26 * 64) / 256, 256, 0, stream>>>(emb_tables, cat_idx, top_in);

  // ---- bottom MLP ----
  gemm_wmma<8><<<dim3(2, 128), 256, 0, stream>>>(xf16, 32, W1t, 32, s1, t1,
                                                 h1, 256, 32, 1);
  gemm_wmma<8><<<dim3(1, 128), 256, 0, stream>>>(h1, 256, W2t, 256, s2, t2,
                                                 h2, 128, 256, 1);
  // layer 3 writes straight into top_in concat slot (cols 351..414)
  gemm_wmma<4><<<dim3(1, 128), 256, 0, stream>>>(h2, 128, W3t, 128, s3, t3,
                                                 top_in + 351, 2080, 128, 0);

  // ---- pairwise interactions (also zeroes pad col 2079) ----
  interact<<<B / 8, 256, 0, stream>>>(top_in);

  // ---- top MLP ----
  gemm_wmma<8><<<dim3(4, 128), 256, 0, stream>>>(top_in, 2080, W4t, 2080,
                                                 s4, t4, h4, 512, 2080, 1);
  gemm_wmma<8><<<dim3(2, 128), 256, 0, stream>>>(h4, 512, W5t, 512, s5, t5,
                                                 h5, 256, 512, 1);
  gemm_wmma<8><<<dim3(1, 128), 256, 0, stream>>>(h5, 256, W6t, 256, s6, t6,
                                                 h6, 128, 256, 1);

  // ---- final scores ----
  final_layer<<<B / 256, 256, 0, stream>>>(h6, W7, b7, (float*)d_out);
}
